// KMeans_82214263980693
// MI455X (gfx1250) — matrix operand order repaired and checked by reading, not compiled, not run on hardware
//
#include <hip/hip_runtime.h>

typedef __attribute__((ext_vector_type(16))) __bf16 v16bf;
typedef __attribute__((ext_vector_type(8)))  float  v8f;
typedef __attribute__((ext_vector_type(2)))  float  v2f;

#define N_PTS  500000
#define DIMS   64
#define KCENT  512
#define NTILES (KCENT / 16)                  // 32 column tiles of 16 centroids
#define BFRAG_BYTES (NTILES * 2 * 2 * 32 * 32)  // 131072: all B fragments

// Workspace layout:
//  [0, 131072)        : B fragments, v16bf[((t*2 + chunk)*2 + part)*32 + lane]
//  [131072, 133120)   : c2s[512] float  (= -0.5 * ||c_k||^2, pre-scaled)

// ---------------------------------------------------------------------------
// Prep: centroids -> bf16 hi/lo B-fragments in contiguous-k B 32x16 layout (same k order as A).
// B element (k_local, n): lane = 16*(k_local/16) + n ; VGPR v holds k_local%16 = 2v, 2v+1.
__global__ __launch_bounds__(256) void prep_b_frags(const float* __restrict__ cent,
                                                    v16bf* __restrict__ gB) {
    int tid  = blockIdx.x * blockDim.x + threadIdx.x;   // 0..2047
    int lane = tid & 31;
    int c    = (tid >> 5) & 1;                          // k-chunk
    int t    = tid >> 6;                                // column tile
    int n    = lane & 15;                               // centroid within tile
    int h    = lane >> 4;                               // k half within chunk
    const float* src = cent + (t * 16 + n) * DIMS + c * 32 + h * 16;
    v16bf hi, lo;
#pragma unroll
    for (int e = 0; e < 16; ++e) {
        float  f  = src[e];
        __bf16 fh = (__bf16)f;
        __bf16 fl = (__bf16)(f - (float)fh);
        hi[e] = fh;
        lo[e] = fl;
    }
    gB[((t * 2 + c) * 2 + 0) * 32 + lane] = hi;
    gB[((t * 2 + c) * 2 + 1) * 32 + lane] = lo;
}

__global__ __launch_bounds__(256) void prep_c2(const float* __restrict__ cent,
                                               float* __restrict__ c2s) {
    int k = blockIdx.x * blockDim.x + threadIdx.x;
    if (k >= KCENT) return;
    const float4* cr = (const float4*)(cent + k * DIMS);
    float s = 0.f;
#pragma unroll
    for (int i = 0; i < DIMS / 4; ++i) {
        float4 v = cr[i];
        s += v.x * v.x + v.y * v.y + v.z * v.z + v.w * v.w;
    }
    c2s[k] = -0.5f * s;   // argmin ||x-c||^2 == argmax (x.c - 0.5*||c||^2)
}

// ---------------------------------------------------------------------------
// Main kernel: 8 waves/block, 32 rows/wave (two 16-row subtiles sharing B).
// All B fragments (128KB) + biases (2KB) staged in LDS once per block; the
// per-tile fragment reads become low-latency ds_load_b128s.
__global__ __launch_bounds__(256) void kmeans_label(const float* __restrict__ x,
                                                    const v16bf* __restrict__ gB,
                                                    const float* __restrict__ gC2s,
                                                    int* __restrict__ out) {
    __shared__ __align__(32) unsigned char smem[BFRAG_BYTES + KCENT * 4];
    v16bf* sB  = (v16bf*)smem;
    float* sC2 = (float*)(smem + BFRAG_BYTES);

    // Cooperative global->LDS stage (uniform trip counts: EXEC stays all-1s)
    {
        const uint4* gsrc = (const uint4*)gB;       // 131072 / 16 = 8192 uint4
        uint4*       sdst = (uint4*)smem;
#pragma unroll
        for (int i = 0; i < BFRAG_BYTES / 16 / 256; ++i)
            sdst[threadIdx.x + i * 256] = gsrc[threadIdx.x + i * 256];
#pragma unroll
        for (int i = 0; i < KCENT / 256; ++i)
            sC2[threadIdx.x + i * 256] = gC2s[threadIdx.x + i * 256];
    }

    const int lane    = threadIdx.x & 31;
    const int wave    = threadIdx.x >> 5;
    const int rowBase = (blockIdx.x * 8 + wave) * 32;
    const int m       = lane & 15;     // A: row-in-tile / C: col-in-tile
    const int h       = lane >> 4;     // lane half

    // --- Build A fragments (contiguous-k A 16x32 layout, same k order as B) while the LDS stage lands:
    // lane L (M = L%16), VGPR v, subword s : k = 16*(L/16) + 2*v + s
    v16bf ah[2][2], al[2][2];   // [subtile][k-chunk], hi and lo parts
#pragma unroll
    for (int sub = 0; sub < 2; ++sub) {
        int row = rowBase + sub * 16 + m;
        row = row < N_PTS ? row : N_PTS - 1;           // clamp; store masked later
        const v2f* xr = (const v2f*)(x + row * DIMS);
#pragma unroll
        for (int c = 0; c < 2; ++c) {
#pragma unroll
            for (int v = 0; v < 8; ++v) {
                int    kb = (h << 4) + (v << 1);
                v2f    f  = __builtin_nontemporal_load(&xr[(c * 32 + kb) >> 1]); // x streamed once: NT
                __bf16 h0 = (__bf16)f.x;
                __bf16 h1 = (__bf16)f.y;
                ah[sub][c][2 * v]     = h0;
                ah[sub][c][2 * v + 1] = h1;
                al[sub][c][2 * v]     = (__bf16)(f.x - (float)h0);
                al[sub][c][2 * v + 1] = (__bf16)(f.y - (float)h1);
            }
        }
    }

    float bestVal[2][8];
    int   bestIdx[2][8];
#pragma unroll
    for (int sub = 0; sub < 2; ++sub)
#pragma unroll
        for (int r = 0; r < 8; ++r) { bestVal[sub][r] = -3.4e38f; bestIdx[sub][r] = 0; }

    __syncthreads();   // LDS stage complete before fragment reads

    for (int t = 0; t < NTILES; ++t) {
        v16bf bh0 = sB[((t * 2 + 0) * 2 + 0) * 32 + lane];
        v16bf bl0 = sB[((t * 2 + 0) * 2 + 1) * 32 + lane];
        v16bf bh1 = sB[((t * 2 + 1) * 2 + 0) * 32 + lane];
        v16bf bl1 = sB[((t * 2 + 1) * 2 + 1) * 32 + lane];
        float c2v = sC2[t * 16 + m];
        int   idx = t * 16 + m;
#pragma unroll
        for (int sub = 0; sub < 2; ++sub) {
            // score = x.c - 0.5*||c||^2 ; bias folded into the accumulator's initial C
            v8f acc = {c2v, c2v, c2v, c2v, c2v, c2v, c2v, c2v};
            // hi*hi + lo*hi + hi*lo  (error-compensated bf16 split of fp32 dot)
            acc = __builtin_amdgcn_wmma_f32_16x16x32_bf16(false, ah[sub][0], false, bh0, (short)0, acc, false, false);
            acc = __builtin_amdgcn_wmma_f32_16x16x32_bf16(false, al[sub][0], false, bh0, (short)0, acc, false, false);
            acc = __builtin_amdgcn_wmma_f32_16x16x32_bf16(false, ah[sub][0], false, bl0, (short)0, acc, false, false);
            acc = __builtin_amdgcn_wmma_f32_16x16x32_bf16(false, ah[sub][1], false, bh1, (short)0, acc, false, false);
            acc = __builtin_amdgcn_wmma_f32_16x16x32_bf16(false, al[sub][1], false, bh1, (short)0, acc, false, false);
            acc = __builtin_amdgcn_wmma_f32_16x16x32_bf16(false, ah[sub][1], false, bl1, (short)0, acc, false, false);
#pragma unroll
            for (int r = 0; r < 8; ++r) {
                if (acc[r] > bestVal[sub][r]) { bestVal[sub][r] = acc[r]; bestIdx[sub][r] = idx; }
            }
        }
    }

    // C layout: VGPR r, lanes 0-15 -> M=r, N=lane ; lanes 16-31 -> M=8+r, N=lane-16.
    // Reduce (max, argmax) across the 16 lanes of each half; xor masks <= 8 stay in-half.
#pragma unroll
    for (int sub = 0; sub < 2; ++sub) {
#pragma unroll
        for (int r = 0; r < 8; ++r) {
            float v = bestVal[sub][r];
            int   i = bestIdx[sub][r];
#pragma unroll
            for (int off = 8; off >= 1; off >>= 1) {
                float ov = __shfl_xor(v, off, 32);
                int   oi = __shfl_xor(i, off, 32);
                if (ov > v || (ov == v && oi < i)) { v = ov; i = oi; }  // first-index tie-break
            }
            if (m == 0) {
                int row = rowBase + sub * 16 + h * 8 + r;
                if (row < N_PTS) out[row] = i;
            }
        }
    }
}

// ---------------------------------------------------------------------------
extern "C" void kernel_launch(void* const* d_in, const int* in_sizes, int n_in,
                              void* d_out, int out_size, void* d_ws, size_t ws_size,
                              hipStream_t stream) {
    const float* x    = (const float*)d_in[0];
    const float* cent = (const float*)d_in[1];
    int*         out  = (int*)d_out;

    v16bf* gB   = (v16bf*)d_ws;
    float* gC2s = (float*)((char*)d_ws + BFRAG_BYTES);

    prep_b_frags<<<8, 256, 0, stream>>>(cent, gB);
    prep_c2<<<2, 256, 0, stream>>>(cent, gC2s);

    int nblocks = (N_PTS + 255) / 256;   // 256 rows per block (8 waves x 32 rows)
    kmeans_label<<<nblocks, 256, 0, stream>>>(x, gB, gC2s, out);
}
